// SigmoidTopKRouter_76845554860187
// MI455X (gfx1250) — matrix operand order repaired and checked by reading, not compiled, not run on hardware
//
#include <hip/hip_runtime.h>
#include <hip/hip_bf16.h>

#define NUM_EXPERTS 64
#define TOPK 8
#define DIM 2048

typedef __attribute__((ext_vector_type(16))) __bf16 v16bf;
typedef __attribute__((ext_vector_type(8)))  float  v8f;

// ---------------------------------------------------------------------------
// Kernel 0: split W_gate fp32 -> bf16 (hi, lo) pair in workspace, and zero the
// global tokens_per_expert accumulator (must be re-zeroed every call).
// ---------------------------------------------------------------------------
__global__ void router_prep(const float* __restrict__ Wg,
                            __bf16* __restrict__ Whi,
                            __bf16* __restrict__ Wlo,
                            float* __restrict__ counts) {
  int i = blockIdx.x * blockDim.x + threadIdx.x;
  if (i < NUM_EXPERTS * DIM) {
    float f = Wg[i];
    __bf16 h = (__bf16)f;
    Whi[i] = h;
    Wlo[i] = (__bf16)(f - (float)h);
  }
  if (blockIdx.x == 0 && threadIdx.x < NUM_EXPERTS) counts[threadIdx.x] = 0.0f;
}

// ---------------------------------------------------------------------------
// Kernel 1: main router. 128 threads = 4 wave32s; each wave computes a
// 32(token) x 64(expert) logit tile via bf16x3 split-precision WMMA with f32
// accumulation, then each thread does top-8 for one token.
// ---------------------------------------------------------------------------
__global__ __launch_bounds__(128)
void router_main(const float* __restrict__ x,
                 const __bf16* __restrict__ Whi,
                 const __bf16* __restrict__ Wlo,
                 const float* __restrict__ bias,
                 float* __restrict__ out_w,
                 float* __restrict__ out_i,
                 float* __restrict__ g_counts) {
  __shared__ float    s_scores[128][NUM_EXPERTS + 1];  // +1 pad vs bank conflicts
  __shared__ float    s_bias[NUM_EXPERTS];
  __shared__ unsigned s_cnt[NUM_EXPERTS];

  const int tid = threadIdx.x;
  if (tid < NUM_EXPERTS) { s_bias[tid] = bias[tid]; s_cnt[tid] = 0u; }

  const int wave = tid >> 5;
  const int lane = tid & 31;
  const int ln   = lane & 15;   // row within 16-row tile (token / expert)
  const int lh   = lane >> 4;   // lane half selects K sub-range
  const int kbA  = lh * 16;     // A layout: lanes 0-15 K=0..15, 16-31 K=16..31
  const int kbB  = lh * 16;     // B layout: lanes 0-15 K=0..15, 16-31 K=16..31

  const int tok_base = blockIdx.x * 128 + wave * 32;
  const float* row0 = x + (size_t)(tok_base + ln) * DIM;
  const float* row1 = row0 + 16 * DIM;

  v8f z = {};
  v8f acc0[4], acc1[4];
#pragma unroll
  for (int t = 0; t < 4; ++t) { acc0[t] = z; acc1[t] = z; }

  // Load 16 fp32 A elements for this lane's K-slots and split into bf16 hi/lo.
  auto loadA = [&](const float* rp, int kk, v16bf& hi, v16bf& lo) {
    const float4 c0 = *(const float4*)(rp + kk + kbA);
    const float4 c1 = *(const float4*)(rp + kk + kbA + 4);
    const float4 c2 = *(const float4*)(rp + kk + kbA + 8);
    const float4 c3 = *(const float4*)(rp + kk + kbA + 12);
    float f[16] = {c0.x, c0.y, c0.z, c0.w, c1.x, c1.y, c1.z, c1.w,
                   c2.x, c2.y, c2.z, c2.w, c3.x, c3.y, c3.z, c3.w};
#pragma unroll
    for (int j = 0; j < 16; ++j) {
      __bf16 h = (__bf16)f[j];
      hi[j] = h;
      lo[j] = (__bf16)(f[j] - (float)h);
    }
  };

  for (int kk = 0; kk < DIM; kk += 32) {
    v16bf a0h, a0l, a1h, a1l;
    loadA(row0, kk, a0h, a0l);
    loadA(row1, kk, a1h, a1l);

    if (kk + 256 < DIM) {  // stream-ahead prefetch for the HBM-bound x stream
      __builtin_prefetch(row0 + kk + 256, 0, 3);
      __builtin_prefetch(row1 + kk + 256, 0, 3);
    }

#pragma unroll
    for (int t = 0; t < 4; ++t) {
      const size_t boff = (size_t)(t * 16 + ln) * DIM + kk + kbB;
      v16bf bh = *(const v16bf*)(Whi + boff);
      v16bf bl = *(const v16bf*)(Wlo + boff);
      // bf16x3: hi*hi + lo*hi + hi*lo, all into f32 accumulators
      acc0[t] = __builtin_amdgcn_wmma_f32_16x16x32_bf16(false, a0h, false, bh, (short)0, acc0[t], false, false);
      acc1[t] = __builtin_amdgcn_wmma_f32_16x16x32_bf16(false, a1h, false, bh, (short)0, acc1[t], false, false);
      acc0[t] = __builtin_amdgcn_wmma_f32_16x16x32_bf16(false, a0l, false, bh, (short)0, acc0[t], false, false);
      acc1[t] = __builtin_amdgcn_wmma_f32_16x16x32_bf16(false, a1l, false, bh, (short)0, acc1[t], false, false);
      acc0[t] = __builtin_amdgcn_wmma_f32_16x16x32_bf16(false, a0h, false, bl, (short)0, acc0[t], false, false);
      acc1[t] = __builtin_amdgcn_wmma_f32_16x16x32_bf16(false, a1h, false, bl, (short)0, acc1[t], false, false);
    }
  }

  // Scatter C tiles to LDS. C layout: VGPR i -> M = i + 8*(lane>=16), N = lane&15.
#pragma unroll
  for (int t = 0; t < 4; ++t) {
#pragma unroll
    for (int i = 0; i < 8; ++i) {
      s_scores[wave * 32 + i + 8 * lh][t * 16 + ln]      = acc0[t][i];
      s_scores[wave * 32 + 16 + i + 8 * lh][t * 16 + ln] = acc1[t][i];
    }
  }
  __syncthreads();

  // --- top-8 per token (one token per thread) ---
  float* row = s_scores[tid];
#pragma unroll 8
  for (int e = 0; e < NUM_EXPERTS; ++e)
    row[e] = 1.0f / (1.0f + __expf(-(row[e] + s_bias[e])));

  float wv[TOPK];
  int   wi[TOPK];
  float wsum = 0.0f;
  for (int k = 0; k < TOPK; ++k) {
    float best = -1.0f;  // sigmoid in (0,1)
    int   bi   = 0;
    for (int e = 0; e < NUM_EXPERTS; ++e) {
      float v = row[e];
      if (v > best) { best = v; bi = e; }  // strict '>' == lowest-index tiebreak
    }
    row[bi] = -2.0f;
    wv[k] = best;
    wi[k] = bi;
    wsum += best;
    atomicAdd(&s_cnt[bi], 1u);
  }
  const float inv = 1.0f / wsum;
  const size_t ob = (size_t)(blockIdx.x * 128 + tid) * TOPK;
#pragma unroll
  for (int k = 0; k < TOPK; ++k) {
    out_w[ob + k] = wv[k] * inv;
    out_i[ob + k] = (float)wi[k];
  }

  __syncthreads();
  if (tid < NUM_EXPERTS) atomicAdd(&g_counts[tid], (float)s_cnt[tid]);
}

// ---------------------------------------------------------------------------
// Kernel 2: EMA + bias update from completed counts.
// ---------------------------------------------------------------------------
__global__ void router_finalize(const float* __restrict__ counts,
                                const float* __restrict__ bias,
                                const float* __restrict__ ema,
                                float* __restrict__ out_ema,
                                float* __restrict__ out_bias,
                                int num_tokens) {
  int e = threadIdx.x;
  if (e < NUM_EXPERTS) {
    float util = counts[e] / ((float)num_tokens * (float)TOPK + 1e-8f);
    float ne = 0.99f * ema[e] + 0.01f * util;
    out_ema[e]  = ne;
    out_bias[e] = bias[e] + 0.001f * (1.0f / (float)NUM_EXPERTS - ne);
  }
}

// ---------------------------------------------------------------------------
extern "C" void kernel_launch(void* const* d_in, const int* in_sizes, int n_in,
                              void* d_out, int out_size, void* d_ws, size_t ws_size,
                              hipStream_t stream) {
  const float* x    = (const float*)d_in[0];
  const float* Wg   = (const float*)d_in[1];
  const float* bias = (const float*)d_in[2];
  const float* ema  = (const float*)d_in[3];
  const int T = in_sizes[0] / DIM;  // 32768

  float* out      = (float*)d_out;
  float* out_w    = out;                        // T*8 weights
  float* out_i    = out + (size_t)T * TOPK;     // T*8 indices (as float)
  float* out_cnt  = out + (size_t)T * TOPK * 2; // 64 tokens_per_expert
  float* out_ema  = out_cnt + NUM_EXPERTS;      // 64
  float* out_bias = out_ema + NUM_EXPERTS;      // 64

  __bf16* Whi = (__bf16*)d_ws;
  __bf16* Wlo = Whi + (size_t)NUM_EXPERTS * DIM;

  router_prep<<<(NUM_EXPERTS * DIM + 255) / 256, 256, 0, stream>>>(Wg, Whi, Wlo, out_cnt);
  router_main<<<T / 128, 128, 0, stream>>>(x, Whi, Wlo, bias, out_w, out_i, out_cnt);
  router_finalize<<<1, 64, 0, stream>>>(out_cnt, bias, ema, out_ema, out_bias, T);
}
